// FP8ProperLinear_51299089384138
// MI455X (gfx1250) — 8-bit matrix operand order repaired and checked by reading, compiled, not run on hardware
//
#include <hip/hip_runtime.h>
#include <stdint.h>

#define IN_F   4096
#define OUT_F  16384
#define MROWS  8192
#define KTILES (IN_F / 128)
#define MAXV   448.0f

typedef int   v16i __attribute__((ext_vector_type(16)));
typedef float v8f  __attribute__((ext_vector_type(8)));
typedef float v4f  __attribute__((ext_vector_type(4)));

__device__ __forceinline__ unsigned f32_to_e4m3_byte(float x) {
    unsigned sgn = (__float_as_uint(x) >> 31) << 7;
    float a = fminf(fabsf(x), MAXV);
    if (!(a > 0.0f)) return sgn;
    int e; float m = frexpf(a, &e);
    int uexp = e - 1;
    if (uexp < -6) {
        int q = (int)rintf(ldexpf(a, 9));
        if (q >= 8) return sgn | 0x08u;
        return sgn | (unsigned)q;
    }
    int mant = (int)rintf(ldexpf(m, 4)) - 8;
    if (mant == 8) { mant = 0; uexp += 1; }
    if (uexp > 8 || (uexp == 8 && mant > 6)) return sgn | 0x7Eu;
    return sgn | ((unsigned)(uexp + 7) << 3) | (unsigned)mant;
}

__device__ __forceinline__ unsigned pack4_e4m3(float a, float b, float c, float d) {
#if __has_builtin(__builtin_amdgcn_cvt_pk_fp8_f32)
    int w = 0;
    w = __builtin_amdgcn_cvt_pk_fp8_f32(a, b, w, false);
    w = __builtin_amdgcn_cvt_pk_fp8_f32(c, d, w, true);
    return (unsigned)w;
#else
    return f32_to_e4m3_byte(a) | (f32_to_e4m3_byte(b) << 8) |
           (f32_to_e4m3_byte(c) << 16) | (f32_to_e4m3_byte(d) << 24);
#endif
}

__global__ void k_init(unsigned* a) { a[0] = 0u; a[1] = 0u; }

__global__ __launch_bounds__(256) void k_amax(const float* __restrict__ p, size_t n,
                                              unsigned* __restrict__ out) {
    __shared__ float red[256];
    float m = 0.0f;
    size_t stride = (size_t)gridDim.x * blockDim.x;
    for (size_t i = (size_t)blockIdx.x * blockDim.x + threadIdx.x; i < n; i += stride)
        m = fmaxf(m, fminf(fabsf(p[i]), MAXV));
    red[threadIdx.x] = m;
    __syncthreads();
    for (int s = 128; s > 0; s >>= 1) {
        if ((int)threadIdx.x < s) red[threadIdx.x] = fmaxf(red[threadIdx.x], red[threadIdx.x + s]);
        __syncthreads();
    }
    if (threadIdx.x == 0) atomicMax(out, __float_as_uint(red[0]));
}

__global__ __launch_bounds__(256) void k_quant_a(const float* __restrict__ x,
                                                 const unsigned* __restrict__ amax,
                                                 unsigned* __restrict__ A) {
    unsigned id = blockIdx.x * 256u + threadIdx.x;
    float ax = fmaxf(__uint_as_float(amax[0]), 1e-12f);
    float sc = MAXV / ax;
    unsigned t = id >> 9, dw = id & 511u;
    unsigned lane = dw >> 4, d = dw & 15u;
    unsigned mTile = t >> 5, kTile = t & 31u;
    unsigned m = mTile * 16u + (lane & 15u);
    unsigned k = kTile * 128u + ((d >> 3) << 6) + (((d >> 1) & 3u) << 4) + ((d & 1u) << 2) + ((lane >> 4) << 3);
    v4f v = *(const v4f*)(x + (size_t)m * IN_F + k);
    float a0 = fminf(fmaxf(v.x, -MAXV), MAXV) * sc;
    float a1 = fminf(fmaxf(v.y, -MAXV), MAXV) * sc;
    float a2 = fminf(fmaxf(v.z, -MAXV), MAXV) * sc;
    float a3 = fminf(fmaxf(v.w, -MAXV), MAXV) * sc;
    A[id] = pack4_e4m3(a0, a1, a2, a3);
}

__global__ __launch_bounds__(256) void k_quant_b(const float* __restrict__ w,
                                                 const unsigned* __restrict__ amax,
                                                 unsigned* __restrict__ B) {
    unsigned id = blockIdx.x * 256u + threadIdx.x;
    float aw = fmaxf(__uint_as_float(amax[1]), 1e-12f);
    float sc = MAXV / aw;
    unsigned t = id >> 9, dw = id & 511u;
    unsigned lane = dw >> 4, d = dw & 15u;
    unsigned nTile = t >> 5, kTile = t & 31u;
    unsigned n = nTile * 16u + (lane & 15u);
    unsigned k = kTile * 128u + ((d >> 3) << 6) + (((d >> 1) & 3u) << 4) + ((d & 1u) << 2) + ((lane >> 4) << 3);
    v4f v = *(const v4f*)(w + (size_t)n * IN_F + k);
    float a0 = fminf(fmaxf(v.x, -MAXV), MAXV) * sc;
    float a1 = fminf(fmaxf(v.y, -MAXV), MAXV) * sc;
    float a2 = fminf(fmaxf(v.z, -MAXV), MAXV) * sc;
    float a3 = fminf(fmaxf(v.w, -MAXV), MAXV) * sc;
    B[id] = pack4_e4m3(a0, a1, a2, a3);
}

__device__ __forceinline__ unsigned lds_swz(unsigned off) {
    return off ^ (((off >> 8) & 3u) << 4);
}

__device__ __forceinline__ v16i ld_frag(const unsigned char* tile_base, int lane) {
    v16i r;
#pragma unroll
    for (int s = 0; s < 4; ++s) {
        unsigned off = lds_swz((unsigned)lane * 64u + 16u * s);
        int4 q = *(const int4*)(tile_base + off);
        r[4 * s + 0] = q.x; r[4 * s + 1] = q.y; r[4 * s + 2] = q.z; r[4 * s + 3] = q.w;
    }
    return r;
}

__global__ __launch_bounds__(256) void k_gemm(const unsigned char* __restrict__ A,
                                              const unsigned char* __restrict__ B,
                                              const float* __restrict__ bias,
                                              const unsigned* __restrict__ amax,
                                              float* __restrict__ out) {
    __shared__ __align__(64) unsigned char sA[2][8 * 2048];
    __shared__ __align__(64) unsigned char sB[2][8 * 2048];

    const int tid  = threadIdx.x;
    const int lane = tid & 31;
    const int wid  = tid >> 5;
    const int widm = wid & 3;
    const int widn = wid >> 2;
    const int bmT  = blockIdx.y * 8;
    const int bnT  = blockIdx.x * 8;

    auto stage = [&](int buf, int kt) {
#pragma unroll
        for (int c = 0; c < 4; ++c) {
            int chunk = tid + c * 256;
            int tile  = chunk >> 7;
            int off   = (chunk & 127) << 4;
            const unsigned char* ga = A + ((size_t)(bmT + tile) * KTILES + kt) * 2048 + off;
            const unsigned char* gb = B + ((size_t)(bnT + tile) * KTILES + kt) * 2048 + off;
            unsigned la = (unsigned)(uintptr_t)&sA[buf][(tile << 11) + lds_swz((unsigned)off)];
            unsigned lb = (unsigned)(uintptr_t)&sB[buf][(tile << 11) + lds_swz((unsigned)off)];
            asm volatile("global_load_async_to_lds_b128 %0, %1, off" :: "v"(la), "v"(ga) : "memory");
            asm volatile("global_load_async_to_lds_b128 %0, %1, off" :: "v"(lb), "v"(gb) : "memory");
        }
    };

    v8f acc[2][4] = {};

    stage(0, 0);
    asm volatile("s_wait_asynccnt 0" ::: "memory");
    __syncthreads();

    for (int kt = 0; kt < KTILES; ++kt) {
        int cur = kt & 1;
        if (kt + 1 < KTILES) stage(cur ^ 1, kt + 1);

        const v16i a0 = ld_frag(&sA[cur][(widm * 2 + 0) * 2048], lane);
        const v16i a1 = ld_frag(&sA[cur][(widm * 2 + 1) * 2048], lane);
#pragma unroll
        for (int j = 0; j < 4; ++j) {
            const v16i b = ld_frag(&sB[cur][(widn * 4 + j) * 2048], lane);
            acc[0][j] = __builtin_amdgcn_wmma_f32_16x16x128_fp8_fp8(a0, b, (short)0, acc[0][j], false, false);
            acc[1][j] = __builtin_amdgcn_wmma_f32_16x16x128_fp8_fp8(a1, b, (short)0, acc[1][j], false, false);
        }
        asm volatile("s_wait_asynccnt 0" ::: "memory");
        __syncthreads();
    }

    float ax  = fmaxf(__uint_as_float(amax[0]), 1e-12f);
    float aw  = fmaxf(__uint_as_float(amax[1]), 1e-12f);
    float scl = (ax * aw) / (MAXV * MAXV);
    int mbase = blockIdx.y * 128 + widm * 32 + ((lane >> 4) << 3);
    int nbase = blockIdx.x * 128 + widn * 64 + (lane & 15);
#pragma unroll
    for (int i = 0; i < 2; ++i) {
#pragma unroll
        for (int j = 0; j < 4; ++j) {
            float bv = bias[nbase + j * 16];
#pragma unroll
            for (int r = 0; r < 8; ++r) {
                out[(size_t)(mbase + i * 16 + r) * OUT_F + (nbase + j * 16)] = acc[i][j][r] * scl + bv;
            }
        }
    }
}

extern "C" void kernel_launch(void* const* d_in, const int* in_sizes, int n_in,
                              void* d_out, int out_size, void* d_ws, size_t ws_size,
                              hipStream_t stream) {
    (void)in_sizes; (void)n_in; (void)out_size; (void)ws_size;
    const float* x    = (const float*)d_in[0];
    const float* w    = (const float*)d_in[1];
    const float* bias = (const float*)d_in[2];
    float* out = (float*)d_out;

    unsigned char* ws = (unsigned char*)d_ws;
    unsigned* amax = (unsigned*)ws;
    unsigned* Aq   = (unsigned*)(ws + 256);
    unsigned* Bq   = (unsigned*)(ws + 256 + (size_t)MROWS * IN_F);

    k_init<<<1, 1, 0, stream>>>(amax);
    k_amax<<<1024, 256, 0, stream>>>(x, (size_t)MROWS * IN_F, amax + 0);
    k_amax<<<1024, 256, 0, stream>>>(w, (size_t)OUT_F * IN_F, amax + 1);
    k_quant_a<<<(MROWS / 4) * (IN_F / 256), 256, 0, stream>>>(x, amax, Aq);
    k_quant_b<<<(OUT_F / 4) * (IN_F / 256), 256, 0, stream>>>(w, amax, Bq);
    dim3 grid(OUT_F / 128, MROWS / 128);
    k_gemm<<<grid, 256, 0, stream>>>((const unsigned char*)Aq, (const unsigned char*)Bq,
                                     bias, amax, out);
}
